// Net_22093311771330
// MI455X (gfx1250) — matrix operand order repaired and checked by reading, not compiled, not run on hardware
//
#include <hip/hip_runtime.h>
#include <hip/hip_bf16.h>

#define NN      50000
#define MP      50048              // NN padded to multiple of 128 (391 blocks)
#define NE      800000
#define ETOT    (NE + NN)          // edges + self loops
#define HID     128
#define HEADS   8
#define HDIM    16
#define NLAYERS 2
#define BATCH   64
#define INPUT   300
#define KP1     320                // INPUT padded to multiple of 32

typedef __attribute__((ext_vector_type(16))) _Float16 v16h;
typedef __attribute__((ext_vector_type(8)))  float    v8f;

// ---- CDNA5 async global->LDS DMA (ASYNCcnt path) ---------------------------
__device__ __forceinline__ void async_copy16(unsigned lds_off, unsigned gbyte_off,
                                             const void* base) {
    // GVS mode: saddr = 64-bit SGPR base, vaddr = 32-bit per-lane byte offset,
    // vdst = LDS byte address. 16 bytes per lane.
    asm volatile("global_load_async_to_lds_b128 %0, %1, %2"
                 :: "v"(lds_off), "v"(gbyte_off),
                    "s"((unsigned long long)(uintptr_t)base)
                 : "memory");
}
__device__ __forceinline__ void wait_async() {
    asm volatile("s_wait_asynccnt 0" ::: "memory");
}
// ---------------------------------------------------------------------------

__device__ __forceinline__ void atomicMaxF(float* addr, float val) {
    if (val >= 0.0f) atomicMax((int*)addr, __float_as_int(val));
    else             atomicMin((unsigned int*)addr, __float_as_uint(val));
}

__device__ __forceinline__ void edge_sd(const int* __restrict__ ei, int e, int& s, int& d) {
    if (e < NE) { s = ei[e]; d = ei[NE + e]; }
    else        { s = e - NE; d = s; }       // self loop
}

// ---------------------------------------------------------------------------
// Pre-cast / pad kernels (run once per call; outputs deterministic)
// ---------------------------------------------------------------------------
__global__ void cast_embed_kernel(const float* __restrict__ src, _Float16* __restrict__ dst)
{
    int idx = blockIdx.x * blockDim.x + threadIdx.x;
    if (idx >= MP * KP1) return;
    int m = idx / KP1, k = idx - m * KP1;
    float v = (m < NN && k < INPUT) ? src[(size_t)m * INPUT + k] : 0.0f;
    dst[idx] = (_Float16)v;
}

// lin1_W (INPUT x 128) -> f16 transposed [n][k], k padded to KP1
__global__ void cast_w1_kernel(const float* __restrict__ src, _Float16* __restrict__ dst)
{
    int idx = blockIdx.x * blockDim.x + threadIdx.x;
    if (idx >= HID * KP1) return;
    int n = idx / KP1, k = idx - n * KP1;
    float v = (k < INPUT) ? src[(size_t)k * HID + n] : 0.0f;
    dst[idx] = (_Float16)v;
}

// gat_W (2 x 128 x 128) -> f16 transposed per layer [n][k]
__global__ void cast_gw_kernel(const float* __restrict__ src, _Float16* __restrict__ dst)
{
    int idx = blockIdx.x * blockDim.x + threadIdx.x;
    if (idx >= NLAYERS * HID * HID) return;
    int l = idx / (HID * HID);
    int r = idx - l * HID * HID;
    int n = r / HID, k = r - n * HID;
    dst[idx] = (_Float16)src[(size_t)l * HID * HID + (size_t)k * HID + n];
}

// zero the padding rows of x16 once per call
__global__ void zero_x16_tail_kernel(_Float16* __restrict__ x16)
{
    int idx = blockIdx.x * blockDim.x + threadIdx.x;
    if (idx >= (MP - NN) * HID) return;
    x16[(size_t)NN * HID + idx] = (_Float16)0.0f;
}

// ---------------------------------------------------------------------------
// GEMM: C[M x 128] = A16[MP x Kp] * Bt[128 x Kp]^T (+bias), f16 WMMA core.
// A16 row-major padded; Bt is [n][k] (pre-transposed) padded. All staging is
// async DMA into LDS (no guards, no conversion in the hot loop).
// Block = 256 threads (8 waves); block owns 128 rows x 128 cols.
// ---------------------------------------------------------------------------
__global__ __launch_bounds__(256) void gemm128_wmma(
    const _Float16* __restrict__ A16, const _Float16* __restrict__ Bt,
    const float* __restrict__ bias, float* __restrict__ C,
    _Float16* __restrict__ C16, int M, int Kp)
{
    __shared__ alignas(16) _Float16 Alds[128 * 32];   // [m][k], k-stride 32
    __shared__ alignas(16) _Float16 Blds[128 * 32];   // [n][k], k-stride 32

    const int tid  = threadIdx.x;
    const int wave = tid >> 5;
    const int lane = tid & 31;
    const int half = lane >> 4;
    const int lr   = lane & 15;
    const int rowbase = blockIdx.x * 128;

    const unsigned aBase = (unsigned)(uintptr_t)Alds;
    const unsigned bBase = (unsigned)(uintptr_t)Blds;

    v8f acc[8] = {};

    for (int k0 = 0; k0 < Kp; k0 += 32) {
        // async-DMA the 128x32 f16 A tile and 128x32 f16 B tile (8 KB each):
        // 512 chunks of 16B per tile; 2 chunks per thread per tile.
        #pragma unroll
        for (int i = 0; i < 2; ++i) {
            int c = tid + 256 * i;
            int m = c >> 2, part = c & 3;
            unsigned ldsOff = (unsigned)((m * 32 + part * 8) * 2);
            async_copy16(aBase + ldsOff,
                         (unsigned)(((size_t)(rowbase + m) * Kp + k0 + part * 8) * 2),
                         A16);
            async_copy16(bBase + ldsOff,
                         (unsigned)(((size_t)m * Kp + k0 + part * 8) * 2),
                         Bt);
        }
        wait_async();
        __syncthreads();

        // A fragment: 16-bit A 16x32 layout
        union { v16h h; unsigned int u[8]; } af;
        #pragma unroll
        for (int v = 0; v < 8; ++v) {
            int kk = ((v < 4) ? (2 * v) : (16 + 2 * (v - 4))) + 8 * half;
            af.u[v] = *(const unsigned int*)&Alds[(wave * 16 + lr) * 32 + kk];
        }

        #pragma unroll
        for (int nt = 0; nt < 8; ++nt) {
            // B fragment: lane%16 = N col, same K layout as A fragment
            union { v16h h; unsigned int u[8]; } bf;
            #pragma unroll
            for (int v = 0; v < 8; ++v) {
                int kk = ((v < 4) ? (2 * v) : (16 + 2 * (v - 4))) + 8 * half;
                bf.u[v] = *(const unsigned int*)&Blds[(nt * 16 + lr) * 32 + kk];
            }
            acc[nt] = __builtin_amdgcn_wmma_f32_16x16x32_f16(
                false, af.h, false, bf.h, (short)0, acc[nt], false, false);
        }
        __syncthreads();
    }

    // C layout: VGPR v -> M = v (+8 for lanes 16-31), N = lane%16
    #pragma unroll
    for (int nt = 0; nt < 8; ++nt) {
        #pragma unroll
        for (int v = 0; v < 8; ++v) {
            int row = rowbase + wave * 16 + v + 8 * half;
            int col = nt * 16 + lr;
            if (row < M) {
                float r = acc[nt][v];
                if (bias) r += bias[col];
                C[(size_t)row * 128 + col] = r;
                if (C16) C16[(size_t)row * 128 + col] = (_Float16)r;
            }
        }
    }
}

// alpha_src/alpha_dst: per (node, head) 16-wide dot
__global__ void alphas_kernel(const float* __restrict__ h,
                              const float* __restrict__ avs,
                              const float* __restrict__ avd,
                              float* __restrict__ as_, float* __restrict__ ad_)
{
    int idx = blockIdx.x * blockDim.x + threadIdx.x;
    if (idx >= NN * HEADS) return;
    int node = idx >> 3, head = idx & 7;
    const float* hp = h + (size_t)node * HID + head * HDIM;
    float s1 = 0.f, s2 = 0.f;
    #pragma unroll
    for (int d = 0; d < HDIM; ++d) {
        float v = hp[d];
        s1 += v * avs[head * HDIM + d];
        s2 += v * avd[head * HDIM + d];
    }
    as_[idx] = s1;
    ad_[idx] = s2;
}

__global__ void init_kernel(float* __restrict__ out, float* __restrict__ emax,
                            float* __restrict__ denom)
{
    int idx = blockIdx.x * blockDim.x + threadIdx.x;
    if (idx < NN * HID) out[idx] = 0.0f;
    if (idx < NN * HEADS) { emax[idx] = -__builtin_inff(); denom[idx] = 0.0f; }
}

__global__ void edge_max_kernel(const int* __restrict__ ei,
                                const float* __restrict__ as_,
                                const float* __restrict__ ad_,
                                float* __restrict__ emax)
{
    int e = blockIdx.x * blockDim.x + threadIdx.x;
    if (e >= ETOT) return;
    int s, d; edge_sd(ei, e, s, d);
    #pragma unroll
    for (int hh = 0; hh < HEADS; ++hh) {
        float v = as_[s * HEADS + hh] + ad_[d * HEADS + hh];
        v = (v > 0.0f) ? v : 0.2f * v;      // leaky relu
        atomicMaxF(&emax[d * HEADS + hh], v);
    }
}

__global__ void edge_sum_kernel(const int* __restrict__ ei,
                                const float* __restrict__ as_,
                                const float* __restrict__ ad_,
                                const float* __restrict__ emax,
                                float* __restrict__ denom)
{
    int e = blockIdx.x * blockDim.x + threadIdx.x;
    if (e >= ETOT) return;
    int s, d; edge_sd(ei, e, s, d);
    #pragma unroll
    for (int hh = 0; hh < HEADS; ++hh) {
        float v = as_[s * HEADS + hh] + ad_[d * HEADS + hh];
        v = (v > 0.0f) ? v : 0.2f * v;
        atomicAdd(&denom[d * HEADS + hh], __expf(v - emax[d * HEADS + hh]));
    }
}

// one wave per edge; lane owns 4 contiguous features (float4 gather, 4 atomics)
__global__ __launch_bounds__(256) void edge_msg_kernel(
    const int* __restrict__ ei, const float* __restrict__ as_,
    const float* __restrict__ ad_, const float* __restrict__ emax,
    const float* __restrict__ denom, const float* __restrict__ h,
    float* __restrict__ out)
{
    int wave = threadIdx.x >> 5, lane = threadIdx.x & 31;
    int e = blockIdx.x * 8 + wave;
    if (e >= ETOT) return;
    int s, d; edge_sd(ei, e, s, d);
    int head = lane >> 2;                         // 4 lanes per head
    float v = as_[s * HEADS + head] + ad_[d * HEADS + head];
    v = (v > 0.0f) ? v : 0.2f * v;
    float ee    = __expf(v - emax[d * HEADS + head]);
    float alpha = ee / denom[d * HEADS + head];
    const float4 hv = *(const float4*)(h + (size_t)s * HID + lane * 4);
    float* op = out + (size_t)d * HID + lane * 4;
    atomicAdd(op + 0, hv.x * alpha);
    atomicAdd(op + 1, hv.y * alpha);
    atomicAdd(op + 2, hv.z * alpha);
    atomicAdd(op + 3, hv.w * alpha);
}

__global__ void finish_kernel(const float* __restrict__ out,
                              const float* __restrict__ bias,
                              float* __restrict__ x, _Float16* __restrict__ x16)
{
    int idx = blockIdx.x * blockDim.x + threadIdx.x;
    if (idx >= NN * HID) return;
    float v = out[idx] + bias[idx & 127];
    v = (v > 0.0f) ? v : (__expf(v) - 1.0f);      // elu
    x[idx]   = v;
    x16[idx] = (_Float16)v;
}

__global__ void score_kernel(const float* __restrict__ x, const int* __restrict__ tgt,
                             const float* __restrict__ w, const float* __restrict__ b,
                             float* __restrict__ outv)
{
    int t = threadIdx.x;
    if (t >= BATCH) return;
    const float* xp = x + (size_t)tgt[t] * HID;
    float s = 0.f;
    #pragma unroll 4
    for (int f = 0; f < HID; ++f) s += xp[f] * w[f];
    outv[t] = s + b[0];
}

__global__ void loss_kernel(float* __restrict__ outv)
{
    if (threadIdx.x != 0) return;
    float acc = 0.f;
    for (int i = 0; i < BATCH / 2; ++i) {
        float m = 1.0f - (outv[i] - outv[BATCH / 2 + i]);
        acc += (m > 0.0f) ? m : 0.0f;
    }
    outv[BATCH] = acc / (float)(BATCH / 2);
}

extern "C" void kernel_launch(void* const* d_in, const int* in_sizes, int n_in,
                              void* d_out, int out_size, void* d_ws, size_t ws_size,
                              hipStream_t stream)
{
    const float* embed = (const float*)d_in[0];   // (NN, INPUT)
    const int*   ei    = (const int*)  d_in[1];   // (2, NE)
    const int*   tgt   = (const int*)  d_in[2];   // (BATCH,)
    const float* lin1W = (const float*)d_in[4];   // (INPUT, HID)
    const float* lin1b = (const float*)d_in[5];   // (HID,)
    const float* gatW  = (const float*)d_in[6];   // (2, HID, HID)
    const float* aSrc  = (const float*)d_in[7];   // (2, HEADS, HDIM)
    const float* aDst  = (const float*)d_in[8];   // (2, HEADS, HDIM)
    const float* gatB  = (const float*)d_in[9];   // (2, HID)
    const float* lin3W = (const float*)d_in[10];  // (HID, 1)
    const float* lin3b = (const float*)d_in[11];  // (1,)
    float* outp = (float*)d_out;                  // 64 scores + 1 loss

    char*  ws  = (char*)d_ws;
    size_t off = 0;
    auto alloc = [&](size_t bytes) -> void* {
        void* p = (void*)(ws + off);
        off += (bytes + 255) & ~(size_t)255;
        return p;
    };
    float*     x       = (float*)    alloc((size_t)MP * HID * 4);
    float*     h       = (float*)    alloc((size_t)NN * HID * 4);
    float*     oacc    = (float*)    alloc((size_t)NN * HID * 4);
    float*     as_     = (float*)    alloc((size_t)NN * HEADS * 4);
    float*     ad_     = (float*)    alloc((size_t)NN * HEADS * 4);
    float*     emax    = (float*)    alloc((size_t)NN * HEADS * 4);
    float*     denom   = (float*)    alloc((size_t)NN * HEADS * 4);
    _Float16*  embed16 = (_Float16*) alloc((size_t)MP * KP1 * 2);
    _Float16*  x16     = (_Float16*) alloc((size_t)MP * HID * 2);
    _Float16*  w1t     = (_Float16*) alloc((size_t)HID * KP1 * 2);
    _Float16*  gwt     = (_Float16*) alloc((size_t)NLAYERS * HID * HID * 2);
    (void)ws_size; (void)in_sizes; (void)n_in; (void)out_size;

    const dim3 b256(256);
    const int gemm_blocks = MP / 128;             // 391

    // one-time (per call) casts / padding
    cast_embed_kernel<<<(MP * KP1 + 255) / 256, b256, 0, stream>>>(embed, embed16);
    cast_w1_kernel<<<(HID * KP1 + 255) / 256, b256, 0, stream>>>(lin1W, w1t);
    cast_gw_kernel<<<(NLAYERS * HID * HID + 255) / 256, b256, 0, stream>>>(gatW, gwt);
    zero_x16_tail_kernel<<<((MP - NN) * HID + 255) / 256, b256, 0, stream>>>(x16);

    // x = embed @ lin1_W + lin1_b  (WMMA + async DMA staging; K padded to 320)
    gemm128_wmma<<<gemm_blocks, b256, 0, stream>>>(embed16, w1t, lin1b, x, x16, NN, KP1);

    for (int l = 0; l < NLAYERS; ++l) {
        // h = x @ gat_W[l]
        gemm128_wmma<<<gemm_blocks, b256, 0, stream>>>(
            x16, gwt + (size_t)l * HID * HID, nullptr, h, nullptr, NN, HID);
        alphas_kernel<<<(NN * HEADS + 255) / 256, b256, 0, stream>>>(
            h, aSrc + l * HEADS * HDIM, aDst + l * HEADS * HDIM, as_, ad_);
        init_kernel<<<(NN * HID + 255) / 256, b256, 0, stream>>>(oacc, emax, denom);
        edge_max_kernel<<<(ETOT + 255) / 256, b256, 0, stream>>>(ei, as_, ad_, emax);
        edge_sum_kernel<<<(ETOT + 255) / 256, b256, 0, stream>>>(ei, as_, ad_, emax, denom);
        edge_msg_kernel<<<(ETOT + 7) / 8, b256, 0, stream>>>(ei, as_, ad_, emax, denom, h, oacc);
        finish_kernel<<<(NN * HID + 255) / 256, b256, 0, stream>>>(oacc, gatB + l * HID, x, x16);
    }

    score_kernel<<<1, 64, 0, stream>>>(x, tgt, lin3W, lin3b, outp);
    loss_kernel<<<1, 32, 0, stream>>>(outp);
}
